// TimeCondTransformer_83829171684068
// MI455X (gfx1250) — matrix operand order repaired and checked by reading, not compiled, not run on hardware
//
#include <hip/hip_runtime.h>

// ---------------------------------------------------------------------------
// Types for CDNA5 WMMA (gfx1250, wave32)
// ---------------------------------------------------------------------------
typedef __attribute__((ext_vector_type(8)))  __bf16 v8bf;
typedef __attribute__((ext_vector_type(16))) __bf16 v16bf;
typedef __attribute__((ext_vector_type(8)))  float  v8f;

#define DEV __device__ __forceinline__

// A-fragment (16x32 bf16, MxK): lane L<16 holds row M=L, K = {off..off+7, off+16..off+23}
// Caller passes p already offset by (lane>=16 ? 8 : 0) elements.
DEV v16bf ldfragA(const __bf16* p) {
  v8bf lo = *(const v8bf*)(p);
  v8bf hi = *(const v8bf*)(p + 16);
  return __builtin_shufflevector(lo, hi, 0,1,2,3,4,5,6,7,8,9,10,11,12,13,14,15);
}
// B-fragment (32x16 bf16, KxN) built from row-major W^T (N x K): lane L<16 holds
// col N=L, K = {off..off+7, off+16..off+23}. Caller passes p already offset by (lane>=16 ? 8 : 0).
DEV v16bf ldfragB(const __bf16* p) {
  v8bf lo = *(const v8bf*)(p);
  v8bf hi = *(const v8bf*)(p + 16);
  return __builtin_shufflevector(lo, hi, 0,1,2,3,4,5,6,7,8,9,10,11,12,13,14,15);
}
DEV v8f wmma_bf16(v16bf a, v16bf b, v8f c) {
  return __builtin_amdgcn_wmma_f32_16x16x32_bf16(false, a, false, b, (short)0, c,
                                                 false, false);
}
DEV float sigmoidf_(float v) { return 1.f / (1.f + __expf(-v)); }

// ---------------------------------------------------------------------------
// Constants
// ---------------------------------------------------------------------------
#define Bsz   4
#define Nseq  1024
#define DIM   512
#define HEADS 8
#define DH    64
#define HID   512
#define INNER 2048
#define TCD   512
#define BN    (Bsz * Nseq)
#define SCALE 0.125f

// ---------------------------------------------------------------------------
// Weight convert + transpose: W (K x N, f32 row-major) -> W^T (N x K, bf16)
// ---------------------------------------------------------------------------
__global__ void __launch_bounds__(256) k_convT(const float* __restrict__ W,
                                               __bf16* __restrict__ WT,
                                               int K, int N) {
  size_t idx = (size_t)blockIdx.x * 256 + threadIdx.x;
  if (idx >= (size_t)K * N) return;
  int kq = (int)(idx / N), n = (int)(idx % N);
  WT[(size_t)n * K + kq] = (__bf16)W[idx];
}

// x_work = x * mask  (fp32)
__global__ void __launch_bounds__(256) k_initx(const float* __restrict__ x,
                                               const float* __restrict__ mask,
                                               float* __restrict__ xw) {
  size_t e = (size_t)blockIdx.x * 256 + threadIdx.x;
  xw[e] = x[e] * mask[e >> 9];
}
__global__ void __launch_bounds__(256) k_copyout(const float* __restrict__ xw,
                                                 float* __restrict__ out) {
  size_t e = (size_t)blockIdx.x * 256 + threadIdx.x;
  out[e] = xw[e];
}

// ---------------------------------------------------------------------------
// Time conditioning: sc_sh = silu(time) @ W + b ; split into sc / sh
// One thread per output column (B * 2*Ncols threads), 512-long dot each.
// ---------------------------------------------------------------------------
__global__ void __launch_bounds__(256) k_timecond(const float* __restrict__ timep,
                                                  const float* __restrict__ W,
                                                  const float* __restrict__ bias,
                                                  float* __restrict__ sc,
                                                  float* __restrict__ sh,
                                                  int Ncols) {
  int o  = blockIdx.x * 256 + threadIdx.x;
  int b  = o / (2 * Ncols);
  int c2 = o % (2 * Ncols);
  const float* tp = timep + (size_t)b * TCD;
  float acc = bias[c2];
  for (int t = 0; t < TCD; ++t) {
    float tv = tp[t];
    acc += (tv * sigmoidf_(tv)) * W[(size_t)t * (2 * Ncols) + c2];
  }
  if (c2 < Ncols) sc[b * Ncols + c2] = acc;
  else            sh[b * Ncols + (c2 - Ncols)] = acc;
}

// ---------------------------------------------------------------------------
// LayerNorm (+ optional scale/shift modulate + mask) -> bf16; optional bf16 copy
// of x. One block (256 thr) per row of 512.
// ---------------------------------------------------------------------------
__global__ void __launch_bounds__(256) k_ln_mod(const float* __restrict__ x,
                                                const float* __restrict__ g,
                                                const float* __restrict__ sc,
                                                const float* __restrict__ sh,
                                                const float* __restrict__ mask,
                                                __bf16* __restrict__ xn,
                                                __bf16* __restrict__ xb,
                                                int modFlag) {
  __shared__ float red[16];
  int row = blockIdx.x;        // 0..BN-1  (== b*Nseq + i)
  int b   = row >> 10;
  const float* xr = x + (size_t)row * DIM;
  int t = threadIdx.x;
  float v0 = xr[t], v1 = xr[t + 256];
  float s = v0 + v1, sq = v0 * v0 + v1 * v1;
  for (int off = 16; off; off >>= 1) {
    s  += __shfl_down(s,  off, 32);
    sq += __shfl_down(sq, off, 32);
  }
  int wave = t >> 5, lane = t & 31;
  if (lane == 0) { red[wave] = s; red[wave + 8] = sq; }
  __syncthreads();
  if (t == 0) {
    float ts = 0.f, tq = 0.f;
    for (int i = 0; i < 8; ++i) { ts += red[i]; tq += red[i + 8]; }
    red[0] = ts; red[8] = tq;
  }
  __syncthreads();
  float mean = red[0] * (1.f / DIM);
  float var  = red[8] * (1.f / DIM) - mean * mean;
  float rstd = rsqrtf(var + 1e-5f);
  float mk   = mask[row];
#pragma unroll
  for (int e = 0; e < 2; ++e) {
    int   c  = t + e * 256;
    float xv = (e == 0) ? v0 : v1;
    float ln = (xv - mean) * rstd * g[c];
    if (modFlag) ln = (ln * (sc[b * DIM + c] + 1.f) + sh[b * DIM + c]) * mk;
    xn[(size_t)row * DIM + c] = (__bf16)ln;
    if (xb) xb[(size_t)row * DIM + c] = (__bf16)xv;
  }
}

// ---------------------------------------------------------------------------
// WMMA GEMM: C(MxN) = A(MxK, bf16) @ B(KxN) with B given as B^T (NxK, bf16).
// Block = 256 thr = 8 waves arranged 2(M) x 4(N); each wave owns a 64x32 tile
// (4x2 grid of v_wmma_f32_16x16x32_bf16) -> block tile 128 x 128.
// K-loop is explicitly double-buffered over 64-K chunks (K % 64 == 0 for all
// call sites) so every fragment-load batch hides under 8 WMMAs.
// Epilogue modes:
//  0: outb = bf16(acc * alpha)                      (q projection)
//  1: kv split-store: cols<512 -> k (row-major); cols>=512 -> v^T per (b,h)
//  2: x += (acc + bias?) * mask                     (attn out projection)
//  3: outb = bf16( silu(acc+bias) * (sc+1) + sh )   (ff in)
//  4: x = (x + acc + bias) * mask                   (ff out)
// ---------------------------------------------------------------------------
__global__ void __launch_bounds__(256) k_gemm(const __bf16* __restrict__ A,
                                              const __bf16* __restrict__ BT,
                                              int M, int N, int K,
                                              int mode, float alpha,
                                              float* __restrict__ xacc,
                                              __bf16* __restrict__ outb,
                                              const float* __restrict__ bias,
                                              const float* __restrict__ sc,
                                              const float* __restrict__ sh,
                                              const float* __restrict__ mask,
                                              __bf16* __restrict__ kout,
                                              __bf16* __restrict__ vTout) {
  int lane = threadIdx.x & 31;
  int wave = threadIdx.x >> 5;     // 0..7
  int wm = wave & 1;               // 2 M-subtiles of 64
  int wn = wave >> 1;              // 4 N-subtiles of 32
  int m0 = blockIdx.x * 128 + wm * 64;
  int n0 = blockIdx.y * 128 + wn * 32;
  int l15 = lane & 15, lh = lane >> 4;

  v8f acc[4][2] = {};
  const __bf16* Ap[4];
  const __bf16* Bp[2];
#pragma unroll
  for (int t = 0; t < 4; ++t)
    Ap[t] = A + (size_t)(m0 + t * 16 + l15) * K + lh * 8;
#pragma unroll
  for (int t = 0; t < 2; ++t)
    Bp[t] = BT + (size_t)(n0 + t * 16 + l15) * K + lh * 8;

  v16bf a0[4], b0[2], a1[4], b1[2];
#pragma unroll
  for (int t = 0; t < 4; ++t) a0[t] = ldfragA(Ap[t]);
#pragma unroll
  for (int t = 0; t < 2; ++t) b0[t] = ldfragB(Bp[t]);

  for (int kk = 0; kk < K; kk += 64) {
    // prefetch second half-chunk, then consume the first
#pragma unroll
    for (int t = 0; t < 4; ++t) a1[t] = ldfragA(Ap[t] + kk + 32);
#pragma unroll
    for (int t = 0; t < 2; ++t) b1[t] = ldfragB(Bp[t] + kk + 32);
#pragma unroll
    for (int i = 0; i < 4; ++i)
#pragma unroll
      for (int j = 0; j < 2; ++j)
        acc[i][j] = wmma_bf16(a0[i], b0[j], acc[i][j]);
    // prefetch next chunk's first half, then consume the second
    if (kk + 64 < K) {
#pragma unroll
      for (int t = 0; t < 4; ++t) a0[t] = ldfragA(Ap[t] + kk + 64);
#pragma unroll
      for (int t = 0; t < 2; ++t) b0[t] = ldfragB(Bp[t] + kk + 64);
    }
#pragma unroll
    for (int i = 0; i < 4; ++i)
#pragma unroll
      for (int j = 0; j < 2; ++j)
        acc[i][j] = wmma_bf16(a1[i], b1[j], acc[i][j]);
  }

#pragma unroll
  for (int ti = 0; ti < 4; ++ti)
#pragma unroll
    for (int tj = 0; tj < 2; ++tj)
#pragma unroll
      for (int r = 0; r < 8; ++r) {
        int m = m0 + ti * 16 + r + 8 * lh;
        int n = n0 + tj * 16 + l15;
        float v = acc[ti][tj][r];
        if (mode == 0) {
          outb[(size_t)m * N + n] = (__bf16)(v * alpha);
        } else if (mode == 1) {
          int bb = m >> 10, i = m & 1023;
          if (n < HID) {
            kout[(size_t)m * HID + n] = (__bf16)v;
          } else {
            int c = n - HID, hh = c >> 6, dh = c & 63;
            vTout[((size_t)((bb * HEADS + hh) * DH + dh)) * Nseq + i] = (__bf16)v;
          }
        } else if (mode == 2) {
          if (bias) v += bias[n];
          xacc[(size_t)m * N + n] += v * mask[m];
        } else if (mode == 3) {
          v += bias[n];
          float sv = v * sigmoidf_(v);
          int bb = m >> 10;
          float rr = sv * (sc[bb * N + n] + 1.f) + sh[bb * N + n];
          outb[(size_t)m * N + n] = (__bf16)rr;
        } else { // mode 4
          v += bias[n];
          size_t idx = (size_t)m * N + n;
          xacc[idx] = (xacc[idx] + v) * mask[m];
        }
      }
}

// ---------------------------------------------------------------------------
// Flash attention: one wave owns a 16-row q tile of one (b,h); loops j in
// steps of 32. QK^T and PV via WMMA; online softmax with wave32 shfl
// reductions; P re-staged through LDS (s_wait_dscnt) to become an A fragment.
// Grid: 256 blocks x 256 thr = 2048 waves = B*H*(N/16) tiles.
// ---------------------------------------------------------------------------
__global__ void __launch_bounds__(256) k_attn(const __bf16* __restrict__ q,
                                              const __bf16* __restrict__ k,
                                              const __bf16* __restrict__ vT,
                                              const int*   __restrict__ ri,
                                              const float* __restrict__ relw,
                                              const float* __restrict__ relb,
                                              const float* __restrict__ mask,
                                              __bf16* __restrict__ aout) {
  __shared__ __bf16 psh[8][16 * 32];
  int lane = threadIdx.x & 31;
  int wave = threadIdx.x >> 5;
  int tile = blockIdx.x * 8 + wave;          // 0..2047
  int b  = tile >> 9;
  int h  = (tile >> 6) & 7;
  int i0 = (tile & 63) << 4;
  int l15 = lane & 15, lh = lane >> 4;

  // q A-fragments for K(dh)=0..31 and 32..63 (held across the whole j loop)
  const __bf16* qp = q + ((size_t)(b * Nseq + i0 + l15)) * HID + h * DH + lh * 8;
  v16bf qa0 = ldfragA(qp);
  v16bf qa1 = ldfragA(qp + 32);

  float hb = relb[h];
  int   rii[8];
  float mi[8];
#pragma unroll
  for (int r = 0; r < 8; ++r) {
    int i = i0 + r + 8 * lh;
    rii[r] = ri[b * Nseq + i];
    mi[r]  = mask[b * Nseq + i];
  }

  float m8[8], l8[8];
  v8f o[4] = {};
#pragma unroll
  for (int r = 0; r < 8; ++r) { m8[r] = -3.0e38f; l8[r] = 0.f; }

  __bf16* pbase = &psh[wave][0];

  for (int j0 = 0; j0 < Nseq; j0 += 32) {
    v8f s[2] = {};
#pragma unroll
    for (int t = 0; t < 2; ++t) {
      const __bf16* kp = k + ((size_t)(b * Nseq + j0 + t * 16 + l15)) * HID
                           + h * DH + lh * 8;
      v16bf kb0 = ldfragB(kp);
      v16bf kb1 = ldfragB(kp + 32);
      s[t] = wmma_bf16(qa0, kb0, s[t]);
      s[t] = wmma_bf16(qa1, kb1, s[t]);
    }
    // bias + masking
#pragma unroll
    for (int t = 0; t < 2; ++t) {
      int   j  = j0 + t * 16 + l15;
      int   rj = ri[b * Nseq + j];
      float mj = mask[b * Nseq + j];
#pragma unroll
      for (int r = 0; r < 8; ++r) {
        int d = rii[r] - rj;
        d = d < -32 ? -32 : (d > 32 ? 32 : d);
        float sv = s[t][r] + relw[(d + 32) * HEADS + h] + hb;
        sv -= (1.f - mi[r] * mj) * 1.0e6f;
        s[t][r] = sv;
      }
    }
    // online softmax update (row reductions across the 16-lane halves)
    float fac[8];
#pragma unroll
    for (int r = 0; r < 8; ++r) {
      float v = fmaxf(s[0][r], s[1][r]);
      for (int off = 8; off >= 1; off >>= 1)
        v = fmaxf(v, __shfl_xor(v, off, 16));
      float mnew = fmaxf(m8[r], v);
      fac[r] = __expf(m8[r] - mnew);
      m8[r]  = mnew;
      float p0 = __expf(s[0][r] - mnew);
      float p1 = __expf(s[1][r] - mnew);
      s[0][r] = p0; s[1][r] = p1;
      float rs = p0 + p1;
      for (int off = 8; off >= 1; off >>= 1)
        rs += __shfl_xor(rs, off, 16);
      l8[r] = l8[r] * fac[r] + rs;
    }
#pragma unroll
    for (int t = 0; t < 4; ++t)
#pragma unroll
      for (int r = 0; r < 8; ++r) o[t][r] *= fac[r];

    // stage P (16x32) to LDS row-major, then reload as an A fragment
#pragma unroll
    for (int t = 0; t < 2; ++t)
#pragma unroll
      for (int r = 0; r < 8; ++r) {
        int row = r + 8 * lh, col = t * 16 + l15;
        pbase[row * 32 + col] = (__bf16)s[t][r];
      }
    asm volatile("s_wait_dscnt 0" ::: "memory");
    v16bf pa = ldfragA(pbase + l15 * 32 + lh * 8);

    // PV: o(16x64) += P(16x32) @ V(32x64), V supplied transposed (64 x Nseq)
#pragma unroll
    for (int t = 0; t < 4; ++t) {
      const __bf16* vp = vT + ((size_t)((b * HEADS + h) * DH + t * 16 + l15)) * Nseq
                            + j0 + lh * 8;
      v16bf vb = ldfragB(vp);
      o[t] = wmma_bf16(pa, vb, o[t]);
    }
    asm volatile("s_wait_dscnt 0" ::: "memory");  // protect psh before next iter
  }

  // normalize + store bf16
#pragma unroll
  for (int t = 0; t < 4; ++t)
#pragma unroll
    for (int r = 0; r < 8; ++r) {
      int i  = i0 + r + 8 * lh;
      int dh = t * 16 + l15;
      float ov = o[t][r] / l8[r];
      aout[((size_t)(b * Nseq + i)) * HID + h * DH + dh] = (__bf16)ov;
    }
}

// ---------------------------------------------------------------------------
// Host-side orchestration
// ---------------------------------------------------------------------------
extern "C" void kernel_launch(void* const* d_in, const int* in_sizes, int n_in,
                              void* d_out, int out_size, void* d_ws, size_t ws_size,
                              hipStream_t stream) {
  const float* x    = (const float*)d_in[0];
  const float* timep= (const float*)d_in[1];
  const float* mask = (const float*)d_in[2];
  const int*   ri   = (const int*)  d_in[3];
  const float* relw = (const float*)d_in[4];
  const float* relb = (const float*)d_in[5];
  const float* ang  = (const float*)d_in[6];
  const float* atw  = (const float*)d_in[7];
  const float* atb  = (const float*)d_in[8];
  const float* qw   = (const float*)d_in[9];
  const float* kvw  = (const float*)d_in[10];
  const float* ow   = (const float*)d_in[11];
  const float* fng  = (const float*)d_in[12];
  const float* ftw  = (const float*)d_in[13];
  const float* ftb  = (const float*)d_in[14];
  const float* fiw  = (const float*)d_in[15];
  const float* fib  = (const float*)d_in[16];
  const float* fow  = (const float*)d_in[17];
  const float* fob  = (const float*)d_in[18];

  char* p = (char*)d_ws;
  auto carve = [&](size_t bytes) -> char* {
    char* r = p; p += (bytes + 255) & ~(size_t)255; return r;
  };
  __bf16* wqT  = (__bf16*)carve((size_t)4 * 512 * 512 * 2);
  __bf16* wkvT = (__bf16*)carve((size_t)4 * 1024 * 512 * 2);
  __bf16* woT  = (__bf16*)carve((size_t)4 * 512 * 512 * 2);
  __bf16* wfiT = (__bf16*)carve((size_t)4 * 2048 * 512 * 2);
  __bf16* wfoT = (__bf16*)carve((size_t)4 * 512 * 2048 * 2);
  float*  xw   = (float*) carve((size_t)BN * DIM * 4);
  __bf16* xbf  = (__bf16*)carve((size_t)BN * DIM * 2);
  __bf16* xnbf = (__bf16*)carve((size_t)BN * DIM * 2);
  __bf16* qbf  = (__bf16*)carve((size_t)BN * HID * 2);
  __bf16* kbf  = (__bf16*)carve((size_t)BN * HID * 2);
  __bf16* vT   = (__bf16*)carve((size_t)BN * HID * 2);
  __bf16* abf  = (__bf16*)carve((size_t)BN * HID * 2);
  __bf16* hnbf = (__bf16*)carve((size_t)BN * DIM * 2);
  __bf16* h2bf = (__bf16*)carve((size_t)BN * INNER * 2);
  float*  sc_a = (float*) carve((size_t)Bsz * DIM * 4);
  float*  sh_a = (float*) carve((size_t)Bsz * DIM * 4);
  float*  sc_f = (float*) carve((size_t)Bsz * INNER * 4);
  float*  sh_f = (float*) carve((size_t)Bsz * INNER * 4);

  // --- weight convert+transpose (once per call) ---
  for (int d = 0; d < 4; ++d) {
    k_convT<<<(512 * 512 + 255) / 256, 256, 0, stream>>>(
        qw + (size_t)d * 512 * 512, wqT + (size_t)d * 512 * 512, 512, 512);
    k_convT<<<(512 * 1024 + 255) / 256, 256, 0, stream>>>(
        kvw + (size_t)d * 512 * 1024, wkvT + (size_t)d * 1024 * 512, 512, 1024);
    k_convT<<<(512 * 512 + 255) / 256, 256, 0, stream>>>(
        ow + (size_t)d * 512 * 512, woT + (size_t)d * 512 * 512, 512, 512);
    k_convT<<<(512 * 2048 + 255) / 256, 256, 0, stream>>>(
        fiw + (size_t)d * 512 * 2048, wfiT + (size_t)d * 2048 * 512, 512, 2048);
    k_convT<<<(2048 * 512 + 255) / 256, 256, 0, stream>>>(
        fow + (size_t)d * 2048 * 512, wfoT + (size_t)d * 512 * 2048, 2048, 512);
  }
  k_initx<<<(BN * DIM) / 256, 256, 0, stream>>>(x, mask, xw);

  for (int d = 0; d < 4; ++d) {
    // attention time conditioning
    k_timecond<<<(Bsz * 2 * DIM) / 256, 256, 0, stream>>>(
        timep, atw + (size_t)d * TCD * (2 * DIM), atb + (size_t)d * 2 * DIM,
        sc_a, sh_a, DIM);
    // LN + modulate -> xn (bf16), plus bf16 copy of x for kv
    k_ln_mod<<<BN, 256, 0, stream>>>(xw, ang + (size_t)d * DIM, sc_a, sh_a, mask,
                                     xnbf, xbf, 1);
    // q = xn @ Wq * SCALE
    k_gemm<<<dim3(BN / 128, HID / 128), 256, 0, stream>>>(
        xnbf, wqT + (size_t)d * 512 * 512, BN, HID, DIM, 0, SCALE,
        nullptr, qbf, nullptr, nullptr, nullptr, nullptr, nullptr, nullptr);
    // kv = x @ Wkv  (split store: k row-major, v transposed per (b,h))
    k_gemm<<<dim3(BN / 128, 1024 / 128), 256, 0, stream>>>(
        xbf, wkvT + (size_t)d * 1024 * 512, BN, 1024, DIM, 1, 1.f,
        nullptr, nullptr, nullptr, nullptr, nullptr, nullptr, kbf, vT);
    // flash attention
    k_attn<<<256, 256, 0, stream>>>(qbf, kbf, vT, ri, relw, relb, mask, abf);
    // x += (attn_out @ Wo) * mask
    k_gemm<<<dim3(BN / 128, DIM / 128), 256, 0, stream>>>(
        abf, woT + (size_t)d * 512 * 512, BN, DIM, HID, 2, 1.f,
        xw, nullptr, nullptr, nullptr, nullptr, mask, nullptr, nullptr);
    // FF time conditioning
    k_timecond<<<(Bsz * 2 * INNER) / 256, 256, 0, stream>>>(
        timep, ftw + (size_t)d * TCD * (2 * INNER), ftb + (size_t)d * 2 * INNER,
        sc_f, sh_f, INNER);
    // plain LN (ff)
    k_ln_mod<<<BN, 256, 0, stream>>>(xw, fng + (size_t)d * DIM, nullptr, nullptr,
                                     mask, hnbf, nullptr, 0);
    // h = silu(hn @ Wfi + b) * (sc+1) + sh  -> bf16
    k_gemm<<<dim3(BN / 128, INNER / 128), 256, 0, stream>>>(
        hnbf, wfiT + (size_t)d * 2048 * 512, BN, INNER, DIM, 3, 1.f,
        nullptr, h2bf, fib + (size_t)d * INNER, sc_f, sh_f, nullptr,
        nullptr, nullptr);
    // x = (x + h @ Wfo + b) * mask
    k_gemm<<<dim3(BN / 128, DIM / 128), 256, 0, stream>>>(
        h2bf, wfoT + (size_t)d * 512 * 2048, BN, DIM, INNER, 4, 1.f,
        xw, nullptr, fob + (size_t)d * DIM, nullptr, nullptr, mask,
        nullptr, nullptr);
  }
  k_copyout<<<(BN * DIM) / 256, 256, 0, stream>>>(xw, (float*)d_out);
  (void)in_sizes; (void)n_in; (void)out_size; (void)ws_size;
}
